// GraphNetwork_9577777070291
// MI455X (gfx1250) — matrix operand order repaired and checked by reading, not compiled, not run on hardware
//
#include <hip/hip_runtime.h>
#include <hip/hip_bf16.h>
#include <math.h>

#define NATOMS   65536
#define NEDGES   524288
#define DD       64
#define NLAYERS  4
#define NGRAPHS  2048
#define BN_EPS   1e-3f

#define WAVES           8      // waves per block (wave32)
#define TILES_PER_WAVE  4      // 16-edge tiles per wave
#define EDGES_PER_BLOCK (WAVES * TILES_PER_WAVE * 16)   // 512

typedef __attribute__((ext_vector_type(16))) _Float16 v16h;
typedef __attribute__((ext_vector_type(8)))  float    v8f;

// wave-local LDS RAW fence: same-wave DS ops are in-order; wait for them and
// stop the compiler from reordering LDS traffic across this point.
#define LDS_WAVE_FENCE() asm volatile("s_wait_dscnt 0" ::: "memory")

// -------- WMMA fragment builders (CDNA5 wave32 layouts, CDNA5 ISA) --------
// A (16x32 f16): lane L -> M = L&15, hi = L>>4; element e -> K = hi*8+e (e<8), 16+hi*8+(e-8)
__device__ __forceinline__ v16h load_a_frag(const _Float16* base, int stride, int m, int hi, int kbase) {
    v16h a;
#pragma unroll
    for (int e = 0; e < 16; ++e) {
        int K = (e < 8) ? (hi * 8 + e) : (16 + hi * 8 + (e - 8));
        a[e] = base[m * stride + kbase + K];
    }
    return a;
}
// B (32x16 f16): lane L -> N = n0 + (L&15), hi = L>>4; element e -> K = kbase + (hi*8+e (e<8), 16+hi*8+(e-8)) (row-major weight [K][N])
__device__ __forceinline__ v16h load_b_frag(const _Float16* w, int nstride, int krow, int n0) {
    v16h b;
    const int kb = krow & ~31, ln = krow & 31, h = ln >> 4, n = ln & 15;
#pragma unroll
    for (int e = 0; e < 16; ++e) {
        int K = (e < 8) ? (h * 8 + e) : (16 + h * 8 + (e - 8));
        b[e] = w[(kb + K) * nstride + n0 + n];
    }
    return b;
}

// -------- init kernels --------
__global__ void embed_atoms_kernel(const int* __restrict__ types, const float* __restrict__ emb,
                                   float* __restrict__ out) {
    int idx = blockIdx.x * blockDim.x + threadIdx.x;  // over NATOMS*DD
    out[idx] = emb[types[idx >> 6] * DD + (idx & 63)];
}
__global__ void embed_bonds_kernel(const int* __restrict__ types, const float* __restrict__ emb,
                                   float* __restrict__ out) {
    int idx = blockIdx.x * blockDim.x + threadIdx.x;  // over NEDGES*DD
    out[idx] = emb[types[idx >> 6] * DD + (idx & 63)];
}
__global__ void zero_kernel(float* __restrict__ p) {
    p[blockIdx.x * blockDim.x + threadIdx.x] = 0.0f;
}

// -------- per-layer fused edge kernel: 1 wave == 16 edges/tile, 4 tiles/wave, 8 waves --------
// dynamic LDS (bytes):
//   w1h f16[192*128]=49152 | w2h f16[128*64]=16384 | wah f16[64*64]=8192
//   bn f32[4*64]=1024 | feat f16[8][16*192]=49152 | nb1 f16[8][16*128]=32768
//   sidx int[8][32]=1024                                  total = 157696
#define EDGE_SMEM 157696

__global__ void edge_layer_kernel(int l,
                                  const int*   __restrict__ conn,
                                  const float* __restrict__ atom_in,
                                  float*       __restrict__ atom_out,
                                  float*       __restrict__ bond,
                                  const float* __restrict__ abn_gamma, const float* __restrict__ abn_beta,
                                  const float* __restrict__ abn_mean,  const float* __restrict__ abn_var,
                                  const float* __restrict__ bbn_gamma, const float* __restrict__ bbn_beta,
                                  const float* __restrict__ bbn_mean,  const float* __restrict__ bbn_var,
                                  const float* __restrict__ W1, const float* __restrict__ W2,
                                  const float* __restrict__ b2, const float* __restrict__ Wa) {
    extern __shared__ char smem[];
    _Float16* w1h = (_Float16*)smem;                  // 192*128
    _Float16* w2h = w1h + 192 * 128;                  // 128*64
    _Float16* wah = w2h + 128 * 64;                   // 64*64
    float* a_scale = (float*)(wah + 64 * 64);
    float* a_bias  = a_scale + 64;
    float* b_scale = a_bias + 64;
    float* b_bias  = b_scale + 64;
    _Float16* featAll = (_Float16*)(b_bias + 64);     // WAVES * 16*192
    _Float16* nb1All  = featAll + WAVES * 16 * 192;   // WAVES * 16*128
    int*      sidxAll = (int*)(nb1All + WAVES * 16 * 128);  // WAVES * 32

    const int tid  = threadIdx.x;
    const int wave = tid >> 5;
    const int lane = tid & 31;
    const int m    = lane & 15;
    const int hi   = lane >> 4;

    // ---- convert layer weights to f16 in LDS, fold BN into scale/bias ----
    const float* W1l = W1 + (size_t)l * 192 * 128;
    const float* W2l = W2 + (size_t)l * 128 * 64;
    const float* Wal = Wa + (size_t)l * 64 * 64;
    for (int i = tid; i < 192 * 128; i += blockDim.x) w1h[i] = (_Float16)W1l[i];
    for (int i = tid; i < 128 * 64;  i += blockDim.x) w2h[i] = (_Float16)W2l[i];
    for (int i = tid; i < 64 * 64;   i += blockDim.x) wah[i] = (_Float16)Wal[i];
    if (tid < 64) {
        float s = abn_gamma[l * 64 + tid] * rsqrtf(abn_var[l * 64 + tid] + BN_EPS);
        a_scale[tid] = s;
        a_bias[tid]  = abn_beta[l * 64 + tid] - abn_mean[l * 64 + tid] * s;
        float sb = bbn_gamma[l * 64 + tid] * rsqrtf(bbn_var[l * 64 + tid] + BN_EPS);
        b_scale[tid] = sb;
        b_bias[tid]  = bbn_beta[l * 64 + tid] - bbn_mean[l * 64 + tid] * sb;
    }
    __syncthreads();   // weights/bn are block-shared: one block-wide barrier only

    _Float16* feat = featAll + wave * 16 * 192;
    _Float16* nb1  = nb1All  + wave * 16 * 128;
    int*      ssrc = sidxAll + wave * 32;
    int*      stgt = ssrc + 16;
    const float* b2l = b2 + l * 64;

    for (int t = 0; t < TILES_PER_WAVE; ++t) {
        const int tile = (blockIdx.x * WAVES + wave) * TILES_PER_WAVE + t;
        const int eb   = tile * 16;

        // prefetch next tile's bond rows + connectivity (global_prefetch_b8)
        if (t + 1 < TILES_PER_WAVE) {
            __builtin_prefetch(bond + (size_t)(eb + 16) * 64 + lane * 32, 0, 1);
            if (lane < 16) __builtin_prefetch(conn + (size_t)(eb + 16 + lane) * 2, 0, 1);
        }

        // stage edge endpoints once per tile
        if (lane < 16) {
            ssrc[lane] = conn[(eb + lane) * 2 + 1];   // source
            stgt[lane] = conn[(eb + lane) * 2 + 0];   // target
        }
        LDS_WAVE_FENCE();

        // ---- stage BN'd [src | tgt | bond] (16 x 192) as f16 ----
        for (int row = 0; row < 16; ++row) {
            const int e  = eb + row;
            const int sn = ssrc[row];
            const int tn = stgt[row];
            for (int col = lane; col < 192; col += 32) {
                float v;
                if (col < 128) {
                    int node = (col < 64) ? sn : tn;
                    int f = col & 63;
                    v = atom_in[node * 64 + f] * a_scale[f] + a_bias[f];
                } else {
                    int f = col - 128;
                    v = bond[(size_t)e * 64 + f] * b_scale[f] + b_bias[f];
                }
                feat[row * 192 + col] = (_Float16)v;
            }
        }
        LDS_WAVE_FENCE();

        // ---- GEMM1: (16x192) @ (192x128), tanh ----
        v8f acc[8] = {};
#pragma unroll
        for (int kc = 0; kc < 6; ++kc) {
            v16h a = load_a_frag(feat, 192, m, hi, kc * 32);
#pragma unroll
            for (int nt = 0; nt < 8; ++nt) {
                v16h b = load_b_frag(w1h, 128, kc * 32 + lane, nt * 16);
                acc[nt] = __builtin_amdgcn_wmma_f32_16x16x32_f16(false, a, false, b,
                                                                 (short)0, acc[nt], false, false);
            }
        }
        // C/D layout: vgpr r, lane -> (M = r + 8*hi, N = nt*16 + m)
#pragma unroll
        for (int nt = 0; nt < 8; ++nt)
#pragma unroll
            for (int r = 0; r < 8; ++r)
                nb1[(r + 8 * hi) * 128 + nt * 16 + m] = (_Float16)tanhf(acc[nt][r]);
        LDS_WAVE_FENCE();

        // ---- GEMM2: (16x128) @ (128x64), bias folded into accumulator ----
        v8f acc2[4];
#pragma unroll
        for (int nt = 0; nt < 4; ++nt) {
            float bias = b2l[nt * 16 + m];
#pragma unroll
            for (int r = 0; r < 8; ++r) acc2[nt][r] = bias;
        }
#pragma unroll
        for (int kc = 0; kc < 4; ++kc) {
            v16h a = load_a_frag(nb1, 128, m, hi, kc * 32);
#pragma unroll
            for (int nt = 0; nt < 4; ++nt) {
                v16h b = load_b_frag(w2h, 64, kc * 32 + lane, nt * 16);
                acc2[nt] = __builtin_amdgcn_wmma_f32_16x16x32_f16(false, a, false, b,
                                                                  (short)0, acc2[nt], false, false);
            }
        }
        // ---- GEMM3 (gate): src_bn (16x64) @ Wa (64x64) ----
        v8f acc3[4] = {};
#pragma unroll
        for (int kc = 0; kc < 2; ++kc) {
            v16h a = load_a_frag(feat, 192, m, hi, kc * 32);   // cols 0..63 = source atom
#pragma unroll
            for (int nt = 0; nt < 4; ++nt) {
                v16h b = load_b_frag(wah, 64, kc * 32 + lane, nt * 16);
                acc3[nt] = __builtin_amdgcn_wmma_f32_16x16x32_f16(false, a, false, b,
                                                                  (short)0, acc3[nt], false, false);
            }
        }

        // ---- epilogue: messages (atomic scatter) + bond residual ----
#pragma unroll
        for (int nt = 0; nt < 4; ++nt) {
            int f = nt * 16 + m;
#pragma unroll
            for (int r = 0; r < 8; ++r) {
                int row = r + 8 * hi;
                int e   = eb + row;
                float nbv = acc2[nt][r];
                float sav = tanhf(acc3[nt][r]);
                atomicAdd(&atom_out[(size_t)stgt[row] * 64 + f], sav * nbv);
                bond[(size_t)e * 64 + f] += nbv;   // unique (e,f) per lane: race-free
            }
        }
        LDS_WAVE_FENCE();  // sidx/feat/nb1 reuse next iteration
    }
}

// -------- output head, WMMA: 16 atoms per wave, 8 waves/block --------
// dynamic LDS (bytes): oW1h 8192 | oW2h 4096 | sb1 256 | sb2 128 | sWf 128
//   xh f16[8][16*64]=16384 | h1h f16[8][16*64]=16384 | h2 f32[8][16*32]=16384
#define HEAD_SMEM (8192 + 4096 + 256 + 128 + 128 + 16384 + 16384 + 16384)

__global__ void head_kernel(const float* __restrict__ atom_state,
                            const int*   __restrict__ atom_types,
                            const int*   __restrict__ ngi,
                            const float* __restrict__ oW1, const float* __restrict__ ob1,
                            const float* __restrict__ oW2, const float* __restrict__ ob2,
                            const float* __restrict__ oWf, const float* __restrict__ obf,
                            const float* __restrict__ mean_emb,
                            float* __restrict__ mol) {
    extern __shared__ char smem[];
    _Float16* oW1h = (_Float16*)smem;            // 64*64
    _Float16* oW2h = oW1h + 64 * 64;             // 64*32
    float* sb1 = (float*)(oW2h + 64 * 32);       // 64
    float* sb2 = sb1 + 64;                       // 32
    float* sWf = sb2 + 32;                       // 32
    _Float16* xhAll  = (_Float16*)(sWf + 32);    // WAVES * 16*64
    _Float16* h1hAll = xhAll + WAVES * 16 * 64;  // WAVES * 16*64
    float*    h2All  = (float*)(h1hAll + WAVES * 16 * 64);  // WAVES * 16*32

    const int tid  = threadIdx.x;
    const int wave = tid >> 5;
    const int lane = tid & 31;
    const int m    = lane & 15;
    const int hi   = lane >> 4;

    for (int i = tid; i < 64 * 64; i += blockDim.x) oW1h[i] = (_Float16)oW1[i];
    for (int i = tid; i < 64 * 32; i += blockDim.x) oW2h[i] = (_Float16)oW2[i];
    if (tid < 64) sb1[tid] = ob1[tid];
    if (tid < 32) { sb2[tid] = ob2[tid]; sWf[tid] = oWf[tid]; }
    __syncthreads();

    _Float16* xh  = xhAll  + wave * 16 * 64;
    _Float16* h1h = h1hAll + wave * 16 * 64;
    float*    h2  = h2All  + wave * 16 * 32;
    const int ab = (blockIdx.x * WAVES + wave) * 16;   // atom base

    // stage 16 atoms' state (16x64, contiguous) as f16
    for (int idx = lane; idx < 16 * 64; idx += 32)
        xh[idx] = (_Float16)atom_state[(size_t)ab * 64 + idx];
    LDS_WAVE_FENCE();

    // GEMM1: relu(X @ oW1 + b1)
    v8f acc[4];
#pragma unroll
    for (int nt = 0; nt < 4; ++nt) {
        float bias = sb1[nt * 16 + m];
#pragma unroll
        for (int r = 0; r < 8; ++r) acc[nt][r] = bias;
    }
#pragma unroll
    for (int kc = 0; kc < 2; ++kc) {
        v16h a = load_a_frag(xh, 64, m, hi, kc * 32);
#pragma unroll
        for (int nt = 0; nt < 4; ++nt) {
            v16h b = load_b_frag(oW1h, 64, kc * 32 + lane, nt * 16);
            acc[nt] = __builtin_amdgcn_wmma_f32_16x16x32_f16(false, a, false, b,
                                                             (short)0, acc[nt], false, false);
        }
    }
#pragma unroll
    for (int nt = 0; nt < 4; ++nt)
#pragma unroll
        for (int r = 0; r < 8; ++r)
            h1h[(r + 8 * hi) * 64 + nt * 16 + m] = (_Float16)fmaxf(acc[nt][r], 0.0f);
    LDS_WAVE_FENCE();

    // GEMM2: relu(H1 @ oW2 + b2)
    v8f acc2[2];
#pragma unroll
    for (int nt = 0; nt < 2; ++nt) {
        float bias = sb2[nt * 16 + m];
#pragma unroll
        for (int r = 0; r < 8; ++r) acc2[nt][r] = bias;
    }
#pragma unroll
    for (int kc = 0; kc < 2; ++kc) {
        v16h a = load_a_frag(h1h, 64, m, hi, kc * 32);
#pragma unroll
        for (int nt = 0; nt < 2; ++nt) {
            v16h b = load_b_frag(oW2h, 32, kc * 32 + lane, nt * 16);
            acc2[nt] = __builtin_amdgcn_wmma_f32_16x16x32_f16(false, a, false, b,
                                                              (short)0, acc2[nt], false, false);
        }
    }
#pragma unroll
    for (int nt = 0; nt < 2; ++nt)
#pragma unroll
        for (int r = 0; r < 8; ++r)
            h2[(r + 8 * hi) * 32 + nt * 16 + m] = fmaxf(acc2[nt][r], 0.0f);
    LDS_WAVE_FENCE();

    // final dot(32) + graph reduction
    if (lane < 16) {
        int atom = ab + lane;
        float s = obf[0] + mean_emb[atom_types[atom]];
        for (int j = 0; j < 32; ++j) s += h2[lane * 32 + j] * sWf[j];
        atomicAdd(&mol[ngi[atom]], s);
    }
}

extern "C" void kernel_launch(void* const* d_in, const int* in_sizes, int n_in,
                              void* d_out, int out_size, void* d_ws, size_t ws_size,
                              hipStream_t stream) {
    const int*   atom_types = (const int*)d_in[0];
    const int*   bond_types = (const int*)d_in[1];
    const int*   ngi        = (const int*)d_in[2];
    const int*   conn       = (const int*)d_in[3];
    const float* atom_emb   = (const float*)d_in[4];
    const float* bond_emb   = (const float*)d_in[5];
    const float* atom_mean  = (const float*)d_in[6];
    const float* abn_gamma  = (const float*)d_in[7];
    const float* abn_beta   = (const float*)d_in[8];
    const float* abn_mean   = (const float*)d_in[9];
    const float* abn_var    = (const float*)d_in[10];
    const float* bbn_gamma  = (const float*)d_in[11];
    const float* bbn_beta   = (const float*)d_in[12];
    const float* bbn_mean   = (const float*)d_in[13];
    const float* bbn_var    = (const float*)d_in[14];
    const float* W1         = (const float*)d_in[15];
    const float* W2         = (const float*)d_in[16];
    const float* b2         = (const float*)d_in[17];
    const float* Wa         = (const float*)d_in[18];
    const float* oW1        = (const float*)d_in[19];
    const float* ob1        = (const float*)d_in[20];
    const float* oW2        = (const float*)d_in[21];
    const float* ob2        = (const float*)d_in[22];
    const float* oWf        = (const float*)d_in[23];
    const float* obf        = (const float*)d_in[24];

    float* bond  = (float*)d_ws;                               // NEDGES*DD floats (128 MB)
    float* atomA = bond  + (size_t)NEDGES * DD;                // NATOMS*DD (16 MB)
    float* atomB = atomA + (size_t)NATOMS * DD;                // NATOMS*DD (16 MB)

    embed_atoms_kernel<<<(NATOMS * DD) / 256, 256, 0, stream>>>(atom_types, atom_emb, atomA);
    embed_bonds_kernel<<<(NEDGES * DD) / 256, 256, 0, stream>>>(bond_types, bond_emb, bond);

    for (int l = 0; l < NLAYERS; ++l) {
        float* ain  = (l & 1) ? atomB : atomA;
        float* aout = (l & 1) ? atomA : atomB;
        hipMemcpyAsync(aout, ain, (size_t)NATOMS * DD * sizeof(float),
                       hipMemcpyDeviceToDevice, stream);
        edge_layer_kernel<<<NEDGES / EDGES_PER_BLOCK, 32 * WAVES, EDGE_SMEM, stream>>>(
            l, conn, ain, aout, bond,
            abn_gamma, abn_beta, abn_mean, abn_var,
            bbn_gamma, bbn_beta, bbn_mean, bbn_var,
            W1, W2, b2, Wa);
    }

    zero_kernel<<<NGRAPHS / 256, 256, 0, stream>>>((float*)d_out);
    head_kernel<<<NATOMS / (WAVES * 16), 32 * WAVES, HEAD_SMEM, stream>>>(
        atomA, atom_types, ngi, oW1, ob1, oW2, ob2, oWf, obf, atom_mean, (float*)d_out);
}
